// RecurrentBlock_22591527977089
// MI455X (gfx1250) — matrix operand order repaired and checked by reading, not compiled, not run on hardware
//
#include <hip/hip_runtime.h>
#include <hip/hip_bf16.h>

typedef __attribute__((ext_vector_type(16))) __bf16 v16bf;
typedef __attribute__((ext_vector_type(8)))  float  v8f;

#define B_  8
#define S_  4096
#define D_  1024
#define E_  3
#define EPS_ 1e-5f
#define NC_ 64          // scan chunks
#define CL_ (S_/NC_)    // chunk length = 64

static __device__ __forceinline__ float sigmoidf_(float x) {
    return 1.0f / (1.0f + __expf(-x));
}

// ---------------------------------------------------------------------------
// 1) RMSNorm: x = in * rsqrt(mean(in^2)+eps) * w  -> bf16
__global__ void rmsnorm_bf16_kernel(const float* __restrict__ in,
                                    const float* __restrict__ w,
                                    __bf16* __restrict__ xb) {
    __shared__ float red[256];
    const long base = (long)blockIdx.x * D_;
    float vals[4];
    float ss = 0.f;
    for (int r = 0; r < 4; ++r) {
        int c = threadIdx.x + 256 * r;
        float v = in[base + c];
        vals[r] = v;
        ss += v * v;
    }
    red[threadIdx.x] = ss;
    __syncthreads();
    for (int s = 128; s > 0; s >>= 1) {
        if (threadIdx.x < s) red[threadIdx.x] += red[threadIdx.x + s];
        __syncthreads();
    }
    float scale = rsqrtf(red[0] / (float)D_ + EPS_);
    for (int r = 0; r < 4; ++r) {
        int c = threadIdx.x + 256 * r;
        xb[base + c] = (__bf16)(vals[r] * scale * w[c]);
    }
}

// ---------------------------------------------------------------------------
// 2) mean over S per (b,d)
__global__ void colmean_kernel(const __bf16* __restrict__ xb,
                               float* __restrict__ mean) {
    int idx = blockIdx.x * 256 + threadIdx.x;   // b*D + d
    int b = idx / D_, d = idx % D_;
    const __bf16* p = xb + (long)b * S_ * D_ + d;
    float s = 0.f;
    for (int t = 0; t < S_; ++t) s += (float)p[(long)t * D_];
    mean[idx] = s / (float)S_;
}

// ---------------------------------------------------------------------------
// 3) router: logits = mean @ rw + rb ; probs = softmax ; aux = 0
__global__ void router_kernel(const float* __restrict__ mean,
                              const float* __restrict__ rw,
                              const float* __restrict__ rb,
                              float* __restrict__ probs,
                              float* __restrict__ aux_out) {
    __shared__ float lg[B_][E_];
    int t = threadIdx.x;
    if (t < B_ * E_) {
        int b = t / E_, e = t % E_;
        float s = rb[e];
        for (int d = 0; d < D_; ++d) s += mean[b * D_ + d] * rw[d * E_ + e];
        lg[b][e] = s;
    }
    __syncthreads();
    if (t < B_) {
        float m = fmaxf(lg[t][0], fmaxf(lg[t][1], lg[t][2]));
        float e0 = __expf(lg[t][0] - m);
        float e1 = __expf(lg[t][1] - m);
        float e2 = __expf(lg[t][2] - m);
        float inv = 1.f / (e0 + e1 + e2);
        probs[t * E_ + 0] = e0 * inv;
        probs[t * E_ + 1] = e1 * inv;
        probs[t * E_ + 2] = e2 * inv;
    }
    if (t == 0) *aux_out = 0.f;
}

// ---------------------------------------------------------------------------
// 4) mixed weights, transposed + bf16: dst[b][n][k] = bf16(sum_e p[b][e]*src[e][k][n])
__global__ void mix_transpose_kernel(const float* __restrict__ src,
                                     const float* __restrict__ probs,
                                     __bf16* __restrict__ dst, int N) {
    __shared__ float tile[E_][32][33];
    int k0 = blockIdx.x * 32, n0 = blockIdx.y * 32;
    int tx = threadIdx.x & 31, ty = threadIdx.x >> 5;   // 32 x 8
    for (int e = 0; e < E_; ++e)
        for (int r = 0; r < 4; ++r) {
            int kl = r * 8 + ty;
            tile[e][kl][tx] = src[((long)e * D_ + (k0 + kl)) * N + (n0 + tx)];
        }
    __syncthreads();
    float p[B_][E_];
    for (int b = 0; b < B_; ++b)
        for (int e = 0; e < E_; ++e) p[b][e] = probs[b * E_ + e];
    for (int b = 0; b < B_; ++b)
        for (int r = 0; r < 4; ++r) {
            int nl = r * 8 + ty, kl = tx;
            float v = p[b][0] * tile[0][kl][nl] +
                      p[b][1] * tile[1][kl][nl] +
                      p[b][2] * tile[2][kl][nl];
            dst[((long)b * N + (n0 + nl)) * D_ + (k0 + kl)] = (__bf16)v;
        }
}

// ---------------------------------------------------------------------------
// WMMA fragment helpers (wave32, 16x16x32 bf16)
union Frag128x2 { v16bf v; uint4 q[2]; };

// async copy of one 16B chunk per lane into LDS (gfx1250 ASYNCcnt path)
__device__ __forceinline__ void async_copy16(unsigned lds_byte_addr,
                                             const __bf16* gptr) {
    unsigned long long ga = (unsigned long long)(size_t)gptr;
    asm volatile("global_load_async_to_lds_b128 %0, %1, off"
                 :: "v"(lds_byte_addr), "v"(ga) : "memory");
}
__device__ __forceinline__ void async_wait0() {
    asm volatile("s_wait_asynccnt 0x0" ::: "memory");
}

// B 32x16 fragment from an LDS tile laid out [row][72 k-elems]
__device__ __forceinline__ v16bf lds_b_frag(const __bf16* wtile, int row16,
                                            int kk, int lane) {
    int n = lane & 15, kh = lane >> 4;
    const __bf16* p = wtile + (row16 + n) * 72 + kk + 8 * kh;
    Frag128x2 f;
    f.q[0] = *(const uint4*)(p);
    f.q[1] = *(const uint4*)(p + 16);
    return f.v;
}

// stage 64 rows x 64 k of W^T into an LDS tile (2 async b128 chunks / thread)
// GATE=true maps rows 32..63 to the gate half (row offset +D_ in W^T)
template <bool GATE>
__device__ __forceinline__ void stage_tile(__bf16* dstTile, const __bf16* WT,
                                           int nblk, int ks, int tid) {
#pragma unroll
    for (int i = 0; i < 2; ++i) {
        int c = tid + i * 256;          // 0..511
        int n = c >> 3, ko = (c & 7) * 8;
        int grow = (!GATE || n < 32) ? (nblk + n) : (D_ + nblk + (n - 32));
        async_copy16((unsigned)(size_t)&dstTile[n * 72 + ko],
                     WT + (long)grow * D_ + ks + ko);
    }
}

// one K=32 step: 1 A fragment (global), 4 B fragments (LDS), 4 WMMAs
__device__ __forceinline__ void mm_step4(const __bf16* tile, const __bf16* arow,
                                         int koff, int lane, v8f acc[4]) {
    Frag128x2 fa;
    const __bf16* p = arow + koff;
    fa.q[0] = *(const uint4*)(p);
    fa.q[1] = *(const uint4*)(p + 16);
    v16bf fb0 = lds_b_frag(tile, 0,  koff & 63, lane);
    v16bf fb1 = lds_b_frag(tile, 16, koff & 63, lane);
    v16bf fb2 = lds_b_frag(tile, 32, koff & 63, lane);
    v16bf fb3 = lds_b_frag(tile, 48, koff & 63, lane);
    acc[0] = __builtin_amdgcn_wmma_f32_16x16x32_bf16(false, fa.v, false, fb0, (short)0, acc[0], false, false);
    acc[1] = __builtin_amdgcn_wmma_f32_16x16x32_bf16(false, fa.v, false, fb1, (short)0, acc[1], false, false);
    acc[2] = __builtin_amdgcn_wmma_f32_16x16x32_bf16(false, fa.v, false, fb2, (short)0, acc[2], false, false);
    acc[3] = __builtin_amdgcn_wmma_f32_16x16x32_bf16(false, fa.v, false, fb3, (short)0, acc[3], false, false);
}

// ---------------------------------------------------------------------------
// 5) hg = x @ Wg, epilogue -> a = 1-sigmoid(gate), b = sigmoid(gate)*g(hid)
// 8 waves stacked on M (tile M=128, N=32 hid + 32 gate fused),
// double-buffered LDS W tile staged with async-to-LDS overlapping the WMMAs.
__global__ void gemm_gate_kernel(const __bf16* __restrict__ xb,
                                 const __bf16* __restrict__ wgT,
                                 float* __restrict__ a_arr,
                                 float* __restrict__ b_arr) {
    __shared__ __bf16 wtile[2][64 * 72];
    int tid = threadIdx.x;
    int lane = tid & 31, wid = tid >> 5;
    int b = blockIdx.z;
    int m0 = blockIdx.x * 128 + wid * 16;
    int nblk = blockIdx.y * 32;
    const __bf16* A  = xb  + (long)b * S_ * D_;
    const __bf16* WT = wgT + (long)b * 2 * D_ * D_;
    v8f acc[4] = {{}, {}, {}, {}};      // 0,1: hid subtiles; 2,3: gate subtiles
    int r = lane & 15, kh = lane >> 4;
    const __bf16* arow = A + (long)(m0 + r) * D_ + 8 * kh;

    stage_tile<true>(wtile[0], WT, nblk, 0, tid);
    async_wait0();
    __syncthreads();
    for (int ks = 0; ks < D_; ks += 64) {
        int cb = (ks >> 6) & 1;
        if (ks + 64 < D_)
            stage_tile<true>(wtile[cb ^ 1], WT, nblk, ks + 64, tid);
        mm_step4(wtile[cb], arow + ks, 0,  lane, acc);
        mm_step4(wtile[cb], arow + ks, 32, lane, acc);
        async_wait0();      // own next-slice copies done
        __syncthreads();    // all waves done reading cur & copying next
    }
    int nn = lane & 15, hi = lane >> 4;
    for (int nt = 0; nt < 2; ++nt)
        for (int v = 0; v < 8; ++v) {
            int m = m0 + v + 8 * hi;
            float hid  = acc[nt][v];
            float gate = acc[2 + nt][v];
            float z  = sigmoidf_(gate);
            float gv = (hid >= 0.f) ? (hid + 0.5f) : sigmoidf_(hid);
            long idx = ((long)b * S_ + m) * D_ + nblk + nt * 16 + nn;
            a_arr[idx] = 1.f - z;
            b_arr[idx] = z * gv;
        }
}

// ---------------------------------------------------------------------------
// 6) scan: h[t] = a[t]*h[t-1] + b[t], h[-1] = current_state
__global__ void scan_chunk_kernel(const float* __restrict__ a_arr,
                                  const float* __restrict__ b_arr,
                                  float* __restrict__ Ac, float* __restrict__ Bc) {
    long idx = (long)blockIdx.x * 256 + threadIdx.x;   // over B*NC*D
    int d = idx % D_;
    long r = idx / D_;
    int c = r % NC_;
    int b = r / NC_;
    long base = ((long)b * S_ + c * CL_) * D_ + d;
    float A = 1.f, Bv = 0.f;
    for (int t = 0; t < CL_; ++t) {
        float at = a_arr[base + (long)t * D_];
        float bt = b_arr[base + (long)t * D_];
        A  = at * A;
        Bv = at * Bv + bt;
    }
    Ac[idx] = A;
    Bc[idx] = Bv;
}

__global__ void scan_carry_kernel(const float* __restrict__ Ac,
                                  const float* __restrict__ Bc,
                                  const float* __restrict__ state,
                                  float* __restrict__ cin) {
    int idx = blockIdx.x * 256 + threadIdx.x;   // B*D
    int b = idx / D_, d = idx % D_;
    float carry = state[idx];
    for (int c = 0; c < NC_; ++c) {
        long o = ((long)b * NC_ + c) * D_ + d;
        cin[o] = carry;
        carry = Ac[o] * carry + Bc[o];
    }
}

__global__ void scan_apply_kernel(const float* __restrict__ a_arr,
                                  const float* __restrict__ b_arr,
                                  const float* __restrict__ cin,
                                  __bf16* __restrict__ hb,
                                  float* __restrict__ ns_out) {
    long idx = (long)blockIdx.x * 256 + threadIdx.x;   // over B*NC*D
    int d = idx % D_;
    long r = idx / D_;
    int c = r % NC_;
    int b = r / NC_;
    long base = ((long)b * S_ + c * CL_) * D_ + d;
    float h = cin[((long)b * NC_ + c) * D_ + d];
    for (int t = 0; t < CL_; ++t) {
        float at = a_arr[base + (long)t * D_];
        float bt = b_arr[base + (long)t * D_];
        h = at * h + bt;
        hb[base + (long)t * D_] = (__bf16)h;
    }
    if (c == NC_ - 1) ns_out[b * D_ + d] = h;
}

// ---------------------------------------------------------------------------
// 7) out = h @ Wo + inputs : 8 waves on M (tile M=128, N=64),
// double-buffered async-staged W tile, 4 WMMAs per A fragment.
__global__ void gemm_out_kernel(const __bf16* __restrict__ hb,
                                const __bf16* __restrict__ woT,
                                const float* __restrict__ resid,
                                float* __restrict__ out) {
    __shared__ __bf16 wtile[2][64 * 72];
    int tid = threadIdx.x;
    int lane = tid & 31, wid = tid >> 5;
    int bq = blockIdx.z;
    int m0 = blockIdx.x * 128 + wid * 16;
    int nblk = blockIdx.y * 64;
    const __bf16* A  = hb  + (long)bq * S_ * D_;
    const __bf16* WT = woT + (long)bq * D_ * D_;
    v8f acc[4] = {{}, {}, {}, {}};
    int r = lane & 15, kh = lane >> 4;
    const __bf16* arow = A + (long)(m0 + r) * D_ + 8 * kh;

    stage_tile<false>(wtile[0], WT, nblk, 0, tid);
    async_wait0();
    __syncthreads();
    for (int ks = 0; ks < D_; ks += 64) {
        int cb = (ks >> 6) & 1;
        if (ks + 64 < D_)
            stage_tile<false>(wtile[cb ^ 1], WT, nblk, ks + 64, tid);
        mm_step4(wtile[cb], arow + ks, 0,  lane, acc);
        mm_step4(wtile[cb], arow + ks, 32, lane, acc);
        async_wait0();
        __syncthreads();
    }
    int nn = lane & 15, hi = lane >> 4;
    for (int nt = 0; nt < 4; ++nt)
        for (int v = 0; v < 8; ++v) {
            long idx = ((long)bq * S_ + m0 + v + 8 * hi) * D_ + nblk + nt * 16 + nn;
            out[idx] = acc[nt][v] + resid[idx];
        }
}

// ---------------------------------------------------------------------------
extern "C" void kernel_launch(void* const* d_in, const int* in_sizes, int n_in,
                              void* d_out, int out_size, void* d_ws, size_t ws_size,
                              hipStream_t stream) {
    (void)in_sizes; (void)n_in; (void)out_size; (void)ws_size;
    const float* inputs = (const float*)d_in[0];
    const float* state  = (const float*)d_in[1];
    const float* norm_w = (const float*)d_in[2];
    const float* rw     = (const float*)d_in[3];
    const float* rb     = (const float*)d_in[4];
    const float* w_gh   = (const float*)d_in[5];
    const float* w_out  = (const float*)d_in[6];

    float* out     = (float*)d_out;
    float* ns_out  = out + (size_t)B_ * S_ * D_;
    float* aux_out = ns_out + (size_t)B_ * D_;

    char* ws = (char*)d_ws;
    size_t off = 0;
    auto take = [&](size_t bytes) -> char* {
        char* p = ws + off;
        off += (bytes + 255) & ~(size_t)255;
        return p;
    };
    __bf16* xb    = (__bf16*)take((size_t)B_ * S_ * D_ * 2);
    float*  mean  = (float*) take((size_t)B_ * D_ * 4);
    float*  probs = (float*) take((size_t)B_ * E_ * 4);
    __bf16* wgT   = (__bf16*)take((size_t)B_ * 2 * D_ * D_ * 2);
    __bf16* woT   = (__bf16*)take((size_t)B_ * D_ * D_ * 2);
    float*  aArr  = (float*) take((size_t)B_ * S_ * D_ * 4);
    float*  bArr  = (float*) take((size_t)B_ * S_ * D_ * 4);
    float*  Ac    = (float*) take((size_t)B_ * NC_ * D_ * 4);
    float*  Bc    = (float*) take((size_t)B_ * NC_ * D_ * 4);
    float*  cin   = (float*) take((size_t)B_ * NC_ * D_ * 4);
    __bf16* hbuf  = (__bf16*)take((size_t)B_ * S_ * D_ * 2);

    rmsnorm_bf16_kernel<<<B_ * S_, 256, 0, stream>>>(inputs, norm_w, xb);
    colmean_kernel<<<(B_ * D_) / 256, 256, 0, stream>>>(xb, mean);
    router_kernel<<<1, 256, 0, stream>>>(mean, rw, rb, probs, aux_out);
    mix_transpose_kernel<<<dim3(D_ / 32, (2 * D_) / 32, 1), 256, 0, stream>>>(w_gh, probs, wgT, 2 * D_);
    mix_transpose_kernel<<<dim3(D_ / 32, D_ / 32, 1), 256, 0, stream>>>(w_out, probs, woT, D_);
    gemm_gate_kernel<<<dim3(S_ / 128, D_ / 32, B_), 256, 0, stream>>>(xb, wgT, aArr, bArr);
    scan_chunk_kernel<<<(B_ * NC_ * D_) / 256, 256, 0, stream>>>(aArr, bArr, Ac, Bc);
    scan_carry_kernel<<<(B_ * D_) / 256, 256, 0, stream>>>(Ac, Bc, state, cin);
    scan_apply_kernel<<<(B_ * NC_ * D_) / 256, 256, 0, stream>>>(aArr, bArr, cin, hbuf, ns_out);
    gemm_out_kernel<<<dim3(S_ / 128, D_ / 64, B_), 256, 0, stream>>>(hbuf, woT, inputs, out);
}
